// VisionTransformer_60928406061619
// MI455X (gfx1250) — matrix operand order repaired and checked by reading, not compiled, not run on hardware
//
#include <hip/hip_runtime.h>
#include <hip/hip_bf16.h>

// ======================= ViT-Base forward for gfx1250 =======================
// All big GEMMs use v_wmma_f32_16x16x32_f16 (fp32 in/out, f16 compute, f32
// accumulate).  Staging is branchless float4 -> f16x4 with clamped addresses.
// ===========================================================================

typedef __attribute__((ext_vector_type(16))) _Float16 v16h;
typedef __attribute__((ext_vector_type(4)))  _Float16 v4h;
typedef __attribute__((ext_vector_type(8)))  float    v8f;

#define VB     32
#define VN     197
#define VNP    196
#define VDIM   768
#define VH     12
#define VHD    64
#define VHID   3072
#define VDEPTH 12
#define VCLS   1000
#define VROWS  (VB * VN)          // 6304
#define VSCALE 0.125f             // 64^-0.5

// ---------------- WMMA fragment helpers (ISA 7.12.2 layouts) ----------------
// A fragment 16x32 f16: lanes 0-15 row M=lane hold K {0..7,16..23},
// lanes 16-31 same rows hold K {8..15,24..31}; 2 f16 per VGPR.
__device__ inline v16h load_a_frag(const _Float16* __restrict__ As, int stride,
                                   int row, int kbase, int lane) {
  int r = row + (lane & 15);
  int half = (lane >> 4) & 1;
  v16h f;
#pragma unroll
  for (int jj = 0; jj < 8; ++jj) {
    int k = kbase + ((jj < 4) ? 0 : 16) + half * 8 + 2 * (jj & 3);
    f[2 * jj]     = As[r * stride + k];
    f[2 * jj + 1] = As[r * stride + k + 1];
  }
  return f;
}

// B fragment 32x16 f16 (KxN): lanes 0-15 col N=lane hold K {0..7,16..23},
// lanes 16-31 hold K {8..15,24..31}.
__device__ inline v16h load_b_frag(const _Float16* __restrict__ Bs, int stride,
                                   int col, int kbase, int lane) {
  int c = col + (lane & 15);
  int half = (lane >> 4) & 1;
  v16h f;
#pragma unroll
  for (int jj = 0; jj < 8; ++jj) {
    int k = kbase + ((jj < 4) ? 0 : 16) + half * 8 + 2 * (jj & 3);
    f[2 * jj]     = Bs[k * stride + c];
    f[2 * jj + 1] = Bs[(k + 1) * stride + c];
  }
  return f;
}

__device__ inline v8f wmma_f16(v16h a, v16h b, v8f c) {
  return __builtin_amdgcn_wmma_f32_16x16x32_f16(false, a, false, b,
                                                (short)0, c, false, false);
}

__device__ inline v4h cvt4(float x, float y, float z, float w, bool ok) {
  v4h h;
  h[0] = (_Float16)(ok ? x : 0.f);
  h[1] = (_Float16)(ok ? y : 0.f);
  h[2] = (_Float16)(ok ? z : 0.f);
  h[3] = (_Float16)(ok ? w : 0.f);
  return h;
}

__device__ inline float gelu_exact(float x) {
  return 0.5f * x * (1.0f + erff(x * 0.70710678118654752f));
}

// --------------------------- generic WMMA GEMM ------------------------------
// C[M,N] = act(A[M,K] @ B[K,N] + bias) + residual.  K must be a multiple of
// 64.  256 threads = 8 waves; 128x128x64 workgroup tile; each wave owns a
// 64x32 sub-tile (4x2 accumulator fragments, 16 WMMAs per K-slab).
#define TM 128
#define TN 128
#define TK 64
#define LAS 72    // padded LDS strides (f16); byte strides are 16B-multiples
#define LBS 136

__global__ __launch_bounds__(256) void gemm_f32_wmma_kernel(
    const float* __restrict__ A, int lda,
    const float* __restrict__ B, int ldb,
    float* __restrict__ C, int ldc,
    int M, int N, int K,
    const float* __restrict__ bias,
    const float* __restrict__ residual, int ldr,
    int act) {
  __shared__ __align__(16) _Float16 As[TM][LAS];   // 18.0 KB
  __shared__ __align__(16) _Float16 Bs[TK][LBS];   // 17.0 KB
  int tid = threadIdx.x;
  int lane = tid & 31;
  int wid = tid >> 5;
  int wm = wid >> 2;        // 0..1  (64-row band)
  int wn = wid & 3;         // 0..3  (32-col band)
  int m0 = blockIdx.y * TM;
  int n0 = blockIdx.x * TN;
  v8f acc[4][2] = {};
  for (int k0 = 0; k0 < K; k0 += TK) {
    // stage A tile (128 x 64 fp32 -> f16), branchless: clamp row, zero-select
#pragma unroll
    for (int p = 0; p < 8; ++p) {
      int e = tid + p * 256;              // 2048 float4 packets
      int r = e >> 4, kq = e & 15;
      int gr = m0 + r;
      bool ok = gr < M;
      int grc = ok ? gr : (M - 1);
      const float4 va =
          *(const float4*)(A + (size_t)grc * lda + k0 + kq * 4);
      *(v4h*)(&As[r][kq * 4]) = cvt4(va.x, va.y, va.z, va.w, ok);
    }
    // stage B tile (64 x 128), N is always a multiple of 4 -> uniform guard
#pragma unroll
    for (int p = 0; p < 8; ++p) {
      int e = tid + p * 256;
      int k = e >> 5, cq = e & 31;
      int gc = n0 + cq * 4;
      bool ok = gc < N;
      int gcc = ok ? gc : (N - 4);
      const float4 vb =
          *(const float4*)(B + (size_t)(k0 + k) * ldb + gcc);
      *(v4h*)(&Bs[k][cq * 4]) = cvt4(vb.x, vb.y, vb.z, vb.w, ok);
    }
    // prefetch next A slab (global_prefetch_b8)
    if (k0 + TK < K) {
      int gr = m0 + (tid >> 1);
      if (gr < M) __builtin_prefetch(&A[(size_t)gr * lda + k0 + TK], 0, 1);
    }
    __syncthreads();
#pragma unroll
    for (int kk = 0; kk < TK; kk += 32) {
      v16h af[4], bf[2];
#pragma unroll
      for (int i = 0; i < 4; ++i)
        af[i] = load_a_frag(&As[0][0], LAS, wm * 64 + i * 16, kk, lane);
#pragma unroll
      for (int j = 0; j < 2; ++j)
        bf[j] = load_b_frag(&Bs[0][0], LBS, wn * 32 + j * 16, kk, lane);
#pragma unroll
      for (int i = 0; i < 4; ++i)
#pragma unroll
        for (int j = 0; j < 2; ++j)
          acc[i][j] = wmma_f16(af[i], bf[j], acc[i][j]);
    }
    __syncthreads();
  }
  // epilogue: C/D layout -> row = half*8 + vgpr, col = lane&15
  int half = lane >> 4, cn = lane & 15;
#pragma unroll
  for (int j = 0; j < 2; ++j) {
    int gcol = n0 + wn * 32 + j * 16 + cn;
    if (gcol >= N) continue;
    float bv = bias ? bias[gcol] : 0.f;
#pragma unroll
    for (int i = 0; i < 4; ++i)
#pragma unroll
      for (int r = 0; r < 8; ++r) {
        int grow = m0 + wm * 64 + i * 16 + half * 8 + r;
        if (grow < M) {
          float v = acc[i][j][r] + bv;
          if (act) v = gelu_exact(v);
          if (residual) v += residual[(size_t)grow * ldr + gcol];
          C[(size_t)grow * ldc + gcol] = v;
        }
      }
  }
}

// ------------------------- batched Q.K^T (scores) ---------------------------
// S[b,h,n,m] = SCALE * sum_d Q[b,h,n,d] K[b,h,m,d].  One (b,h) per z-block;
// 64x64 tile; single 64-deep K slab; 8 waves each own 32x16.
__global__ __launch_bounds__(256) void qk_scores_kernel(
    const float* __restrict__ qkv, float* __restrict__ S) {
  __shared__ __align__(16) _Float16 Qs[64][72];
  __shared__ __align__(16) _Float16 Ks[64][72];   // [d][m]
  int tid = threadIdx.x, lane = tid & 31, wid = tid >> 5;
  int wm = wid >> 2, wn = wid & 3;
  int bh = blockIdx.z;
  int b = bh / VH, h = bh - b * VH;
  int n0 = blockIdx.y * 64;
  int m0 = blockIdx.x * 64;
  // stage Q (64 n x 64 d), scale folded in
#pragma unroll
  for (int p = 0; p < 4; ++p) {
    int e = tid + p * 256;                // 1024 float4
    int r = e >> 4, kq = e & 15;
    int n = n0 + r;
    bool ok = n < VN;
    int nc = ok ? n : (VN - 1);
    const float4 v = *(const float4*)(
        qkv + ((size_t)b * VN + nc) * (3 * VDIM) + h * VHD + kq * 4);
    *(v4h*)(&Qs[r][kq * 4]) =
        cvt4(v.x * VSCALE, v.y * VSCALE, v.z * VSCALE, v.w * VSCALE, ok);
  }
  // stage K transposed: load contiguous d, scatter into Ks[d][m]
#pragma unroll
  for (int p = 0; p < 4; ++p) {
    int e = tid + p * 256;
    int c = e >> 4, kq = e & 15;          // c = m index, kq = d quad
    int m = m0 + c;
    bool ok = m < VN;
    int mc = ok ? m : (VN - 1);
    const float4 v = *(const float4*)(
        qkv + ((size_t)b * VN + mc) * (3 * VDIM) + VDIM + h * VHD + kq * 4);
    Ks[kq * 4 + 0][c] = (_Float16)(ok ? v.x : 0.f);
    Ks[kq * 4 + 1][c] = (_Float16)(ok ? v.y : 0.f);
    Ks[kq * 4 + 2][c] = (_Float16)(ok ? v.z : 0.f);
    Ks[kq * 4 + 3][c] = (_Float16)(ok ? v.w : 0.f);
  }
  __syncthreads();
  v8f acc[2] = {};
#pragma unroll
  for (int kk = 0; kk < 64; kk += 32) {
    v16h a0 = load_a_frag(&Qs[0][0], 72, wm * 32, kk, lane);
    v16h a1 = load_a_frag(&Qs[0][0], 72, wm * 32 + 16, kk, lane);
    v16h bf = load_b_frag(&Ks[0][0], 72, wn * 16, kk, lane);
    acc[0] = wmma_f16(a0, bf, acc[0]);
    acc[1] = wmma_f16(a1, bf, acc[1]);
  }
  int half = lane >> 4, cn = lane & 15;
#pragma unroll
  for (int i = 0; i < 2; ++i)
#pragma unroll
    for (int r = 0; r < 8; ++r) {
      int n = n0 + wm * 32 + i * 16 + half * 8 + r;
      int m = m0 + wn * 16 + cn;
      if (n < VN && m < VN)
        S[((size_t)bh * VN + n) * VN + m] = acc[i][r];
    }
}

// ----------------- head-mix -> softmax -> head-mix (per b,n) ----------------
__global__ __launch_bounds__(256) void mix_softmax_mix_kernel(
    const float* __restrict__ S, const float* __restrict__ pre_w,
    const float* __restrict__ post_w, _Float16* __restrict__ P) {
  __shared__ float Sl[VH][200];
  __shared__ float Tl[VH][200];
  __shared__ float red[VH][16];
  __shared__ float gstat[VH];
  __shared__ float wpre[VH * VH], wpost[VH * VH];
  int tid = threadIdx.x;
  int bn = blockIdx.x;
  int b = bn / VN, n = bn - b * VN;
  if (tid < VH * VH) { wpre[tid] = pre_w[tid]; wpost[tid] = post_w[tid]; }
  for (int idx = tid; idx < VH * VN; idx += 256) {
    int hh = idx / VN, m = idx - hh * VN;
    Sl[hh][m] = S[(((size_t)b * VH + hh) * VN + n) * VN + m];
  }
  __syncthreads();
  for (int idx = tid; idx < VH * VN; idx += 256) {
    int g = idx / VN, m = idx - g * VN;
    float acc = 0.f;
#pragma unroll
    for (int hh = 0; hh < VH; ++hh) acc += Sl[hh][m] * wpre[hh * VH + g];
    Tl[g][m] = acc;
  }
  __syncthreads();
  int g = tid >> 4, l = tid & 15;
  if (g < VH) {
    float mx = -3.4e38f;
    for (int m = l; m < VN; m += 16) mx = fmaxf(mx, Tl[g][m]);
    red[g][l] = mx;
  }
  __syncthreads();
  if (g < VH && l == 0) {
    float mx = red[g][0];
    for (int t = 1; t < 16; ++t) mx = fmaxf(mx, red[g][t]);
    gstat[g] = mx;
  }
  __syncthreads();
  if (g < VH) {
    float s = 0.f, mx = gstat[g];
    for (int m = l; m < VN; m += 16) {
      float e = expf(Tl[g][m] - mx);
      Tl[g][m] = e;
      s += e;
    }
    red[g][l] = s;
  }
  __syncthreads();
  if (g < VH && l == 0) {
    float s = 0.f;
    for (int t = 0; t < 16; ++t) s += red[g][t];
    gstat[g] = s;
  }
  __syncthreads();
  for (int idx = tid; idx < VH * VN; idx += 256) {
    int gg = idx / VN, m = idx - gg * VN;
    Tl[gg][m] /= gstat[gg];
  }
  __syncthreads();
  for (int idx = tid; idx < VH * VN; idx += 256) {
    int gp = idx / VN, m = idx - gp * VN;
    float acc = 0.f;
#pragma unroll
    for (int gg = 0; gg < VH; ++gg) acc += Tl[gg][m] * wpost[gg * VH + gp];
    P[(((size_t)b * VH + gp) * VN + n) * VN + m] = (_Float16)acc;
  }
}

// ------------------------- batched P.V (attn output) ------------------------
// O[b,n,h*64+d] = sum_m P[b,h,n,m] V[b,h,m,d]; K=197 with zero-padded tiles.
__global__ __launch_bounds__(256) void attn_v_kernel(
    const _Float16* __restrict__ P, const float* __restrict__ qkv,
    float* __restrict__ O) {
  __shared__ __align__(16) _Float16 Ps[64][40];
  __shared__ __align__(16) _Float16 Vs[32][72];
  int tid = threadIdx.x, lane = tid & 31, wid = tid >> 5;
  int wm = wid >> 2, wn = wid & 3;
  int bh = blockIdx.z;
  int b = bh / VH, h = bh - b * VH;
  int n0 = blockIdx.y * 64;
  v8f acc[2] = {};
  for (int k0 = 0; k0 < VN; k0 += 32) {
    // stage P (64 x 32 f16), branchless scalar loads with clamped address
#pragma unroll
    for (int p = 0; p < 8; ++p) {
      int e = tid + p * 256;
      int r = e >> 5, k = e & 31;
      int n = n0 + r, m = k0 + k;
      bool ok = (n < VN) && (m < VN);
      int nc = n < VN ? n : (VN - 1);
      int mc = m < VN ? m : (VN - 1);
      _Float16 v = P[((size_t)bh * VN + nc) * VN + mc];
      Ps[r][k] = ok ? v : (_Float16)0.f;
    }
    // stage V (32 m x 64 d)
#pragma unroll
    for (int p = 0; p < 2; ++p) {
      int e = tid + p * 256;                // 512 float4
      int k = e >> 4, kq = e & 15;
      int m = k0 + k;
      bool ok = m < VN;
      int mc = ok ? m : (VN - 1);
      const float4 v = *(const float4*)(
          qkv + ((size_t)b * VN + mc) * (3 * VDIM) + 2 * VDIM + h * VHD +
          kq * 4);
      *(v4h*)(&Vs[k][kq * 4]) = cvt4(v.x, v.y, v.z, v.w, ok);
    }
    __syncthreads();
    v16h a0 = load_a_frag(&Ps[0][0], 40, wm * 32, 0, lane);
    v16h a1 = load_a_frag(&Ps[0][0], 40, wm * 32 + 16, 0, lane);
    v16h bf = load_b_frag(&Vs[0][0], 72, wn * 16, 0, lane);
    acc[0] = wmma_f16(a0, bf, acc[0]);
    acc[1] = wmma_f16(a1, bf, acc[1]);
    __syncthreads();
  }
  int half = lane >> 4, cn = lane & 15;
#pragma unroll
  for (int i = 0; i < 2; ++i)
#pragma unroll
    for (int r = 0; r < 8; ++r) {
      int n = n0 + wm * 32 + i * 16 + half * 8 + r;
      int d = wn * 16 + cn;
      if (n < VN)
        O[((size_t)b * VN + n) * VDIM + h * VHD + d] = acc[i][r];
    }
}

// ------------------------------- layernorm ----------------------------------
__global__ __launch_bounds__(256) void layernorm_kernel(
    const float* __restrict__ X, int xstride,
    float* __restrict__ Y, int ystride,
    const float* __restrict__ w, const float* __restrict__ bias) {
  __shared__ float rs[256], rss[256];
  int r = blockIdx.x, tid = threadIdx.x;
  const float* x = X + (size_t)r * xstride;
  float s = 0.f, ss = 0.f;
  for (int i = tid; i < VDIM; i += 256) {
    float v = x[i];
    s += v;
    ss += v * v;
  }
  rs[tid] = s;
  rss[tid] = ss;
  __syncthreads();
  for (int off = 128; off > 0; off >>= 1) {
    if (tid < off) { rs[tid] += rs[tid + off]; rss[tid] += rss[tid + off]; }
    __syncthreads();
  }
  float mean = rs[0] * (1.f / VDIM);
  float var = rss[0] * (1.f / VDIM) - mean * mean;
  float inv = rsqrtf(var + 1e-6f);
  float* y = Y + (size_t)r * ystride;
  for (int i = tid; i < VDIM; i += 256)
    y[i] = (x[i] - mean) * inv * w[i] + bias[i];
}

// ------------------------------ im2col (patches) ----------------------------
__global__ __launch_bounds__(256) void im2col_kernel(
    const float* __restrict__ x, float* __restrict__ out) {
  int idx = blockIdx.x * 256 + threadIdx.x;
  if (idx >= VB * VNP * VDIM) return;
  int col = idx % VDIM;
  int row = idx / VDIM;
  int b = row / VNP, p = row - b * VNP;
  int gy = p / 14, gx = p - gy * 14;
  int c = col >> 8, rem = col & 255;
  int py = rem >> 4, px = rem & 15;
  out[idx] = x[(((size_t)b * 3 + c) * 224 + gy * 16 + py) * 224 + gx * 16 + px];
}

// --------------------- assemble h = [cls | patch] + pos ---------------------
__global__ __launch_bounds__(256) void assemble_h_kernel(
    const float* __restrict__ pe, const float* __restrict__ cls,
    const float* __restrict__ pos, float* __restrict__ h) {
  int idx = blockIdx.x * 256 + threadIdx.x;
  if (idx >= VROWS * VDIM) return;
  int col = idx % VDIM;
  int row = idx / VDIM;
  int b = row / VN, n = row - b * VN;
  float v = (n == 0) ? cls[col] : pe[((size_t)b * VNP + (n - 1)) * VDIM + col];
  h[idx] = v + pos[n * VDIM + col];
}

// ---------------- CDNA5 async / TDM probe kernels (never launched) ----------
// Compile-only probes for the gfx1250 async-copy and Tensor Data Mover paths.
typedef __attribute__((ext_vector_type(4))) unsigned int pv4u;
typedef __attribute__((ext_vector_type(8))) int pv8i;
typedef __attribute__((ext_vector_type(4))) int pv4i;

__global__ __launch_bounds__(32) void cdna5_async_probe_kernel(
    const float* __restrict__ g, float* __restrict__ o) {
  __shared__ float buf[128];
  unsigned lds_off = threadIdx.x * 16u;   // buf is the only LDS object (off 0)
  unsigned long long ga =
      (unsigned long long)(uintptr_t)g + (unsigned long long)threadIdx.x * 16u;
  asm volatile("global_load_async_to_lds_b128 %0, %1, off"
               :: "v"(lds_off), "v"(ga) : "memory");
  asm volatile("s_wait_asynccnt 0" ::: "memory");
  __syncthreads();
  o[threadIdx.x] = buf[threadIdx.x];
}

__global__ __launch_bounds__(32) void cdna5_tdm_probe_kernel(
    float* __restrict__ o) {
  __shared__ float buf[128];
#if __has_builtin(__builtin_amdgcn_tensor_load_to_lds)
  pv4u g0 = {0u, 0u, 0u, 0u};
  pv8i g1 = {0, 0, 0, 0, 0, 0, 0, 0};
  pv4i g2 = {0, 0, 0, 0};
  pv4i g3 = {0, 0, 0, 0};
#if __clang_major__ >= 23
  __builtin_amdgcn_tensor_load_to_lds(g0, g1, g2, g3, g1, 0);
#else
  __builtin_amdgcn_tensor_load_to_lds(g0, g1, g2, g3, 0);
#endif
#endif
#if __has_builtin(__builtin_amdgcn_s_wait_tensorcnt)
  __builtin_amdgcn_s_wait_tensorcnt(0);
#endif
  __syncthreads();
  o[threadIdx.x] = buf[threadIdx.x];
}

// ============================== host orchestration ==========================
extern "C" void kernel_launch(void* const* d_in, const int* in_sizes, int n_in,
                              void* d_out, int out_size, void* d_ws,
                              size_t ws_size, hipStream_t stream) {
  (void)in_sizes; (void)n_in; (void)out_size; (void)ws_size;
  const float* x         = (const float*)d_in[0];
  const float* patch_w   = (const float*)d_in[1];
  const float* patch_b   = (const float*)d_in[2];
  const float* cls_tok   = (const float*)d_in[3];
  const float* pos_emb   = (const float*)d_in[4];
  const float* ln1_w     = (const float*)d_in[5];
  const float* ln1_b     = (const float*)d_in[6];
  const float* qkv_w     = (const float*)d_in[7];
  const float* qkv_b     = (const float*)d_in[8];
  const float* pre_w     = (const float*)d_in[9];
  const float* post_w    = (const float*)d_in[10];
  const float* aproj_w   = (const float*)d_in[11];
  const float* aproj_b   = (const float*)d_in[12];
  const float* ln2_w     = (const float*)d_in[13];
  const float* ln2_b     = (const float*)d_in[14];
  const float* fc1_w     = (const float*)d_in[15];
  const float* fc1_b     = (const float*)d_in[16];
  const float* fc2_w     = (const float*)d_in[17];
  const float* fc2_b     = (const float*)d_in[18];
  const float* norm_w    = (const float*)d_in[19];
  const float* norm_b    = (const float*)d_in[20];
  const float* head_w    = (const float*)d_in[21];
  const float* head_b    = (const float*)d_in[22];
  float* out = (float*)d_out;

  // workspace layout (~206 MB, all offsets 256B-aligned)
  const size_t SZ_H = (size_t)VROWS * VDIM * sizeof(float);      // h
  const size_t SZ_A = SZ_H;                                      // a / o / im2col
  const size_t SZ_T = (size_t)VROWS * VHID * sizeof(float);      // qkv / mlp
  const size_t SZ_S = (size_t)VB * VH * VN * VN * sizeof(float); // scores
  char* ws = (char*)d_ws;
  float*    hb = (float*)ws;
  float*    ab = (float*)(ws + SZ_H);
  float*    tb = (float*)(ws + SZ_H + SZ_A);
  float*    Sb = (float*)(ws + SZ_H + SZ_A + SZ_T);
  _Float16* Pb = (_Float16*)(ws + SZ_H + SZ_A + SZ_T + SZ_S);

  // ---- patch embedding ----
  im2col_kernel<<<(VB * VNP * VDIM + 255) / 256, 256, 0, stream>>>(x, ab);
  gemm_f32_wmma_kernel<<<dim3(6, 49), 256, 0, stream>>>(
      ab, VDIM, patch_w, VDIM, tb, VDIM, VB * VNP, VDIM, VDIM,
      patch_b, nullptr, 0, 0);
  assemble_h_kernel<<<(VROWS * VDIM + 255) / 256, 256, 0, stream>>>(
      tb, cls_tok, pos_emb, hb);

  // ---- transformer layers ----
  for (int i = 0; i < VDEPTH; ++i) {
    layernorm_kernel<<<VROWS, 256, 0, stream>>>(
        hb, VDIM, ab, VDIM, ln1_w + i * VDIM, ln1_b + i * VDIM);
    gemm_f32_wmma_kernel<<<dim3(18, 50), 256, 0, stream>>>(
        ab, VDIM, qkv_w + (size_t)i * VDIM * 3 * VDIM, 3 * VDIM,
        tb, 3 * VDIM, VROWS, 3 * VDIM, VDIM,
        qkv_b + i * 3 * VDIM, nullptr, 0, 0);
    qk_scores_kernel<<<dim3(4, 4, VB * VH), 256, 0, stream>>>(tb, Sb);
    mix_softmax_mix_kernel<<<VROWS, 256, 0, stream>>>(
        Sb, pre_w + i * VH * VH, post_w + i * VH * VH, Pb);
    attn_v_kernel<<<dim3(1, 4, VB * VH), 256, 0, stream>>>(Pb, tb, ab);
    gemm_f32_wmma_kernel<<<dim3(6, 50), 256, 0, stream>>>(
        ab, VDIM, aproj_w + (size_t)i * VDIM * VDIM, VDIM,
        hb, VDIM, VROWS, VDIM, VDIM,
        aproj_b + i * VDIM, hb, VDIM, 0);
    layernorm_kernel<<<VROWS, 256, 0, stream>>>(
        hb, VDIM, ab, VDIM, ln2_w + i * VDIM, ln2_b + i * VDIM);
    gemm_f32_wmma_kernel<<<dim3(24, 50), 256, 0, stream>>>(
        ab, VDIM, fc1_w + (size_t)i * VDIM * VHID, VHID,
        tb, VHID, VROWS, VHID, VDIM,
        fc1_b + i * VHID, nullptr, 0, /*gelu=*/1);
    gemm_f32_wmma_kernel<<<dim3(6, 50), 256, 0, stream>>>(
        tb, VHID, fc2_w + (size_t)i * VHID * VDIM, VDIM,
        hb, VDIM, VROWS, VDIM, VHID,
        fc2_b + i * VDIM, hb, VDIM, 0);
  }

  // ---- final norm (cls rows only, strided) + classifier head ----
  layernorm_kernel<<<VB, 256, 0, stream>>>(
      hb, VN * VDIM, ab, VDIM, norm_w, norm_b);
  gemm_f32_wmma_kernel<<<dim3(8, 1), 256, 0, stream>>>(
      ab, VDIM, head_w, VCLS, out, VCLS, VB, VCLS, VDIM,
      head_b, nullptr, 0, 0);
}
